// GLM4MoEWrapper_35021163332174
// MI455X (gfx1250) — matrix operand order repaired and checked by reading, not compiled, not run on hardware
//
#include <hip/hip_runtime.h>

#define HDIM 1024
#define IDIM 2048
#define NEXP 8
#define TTOK 4096
#define LS 40   // LDS row stride in ushort elems: 80B = 16B-aligned rows, 20-bank skew -> conflict-free

typedef __attribute__((ext_vector_type(16))) __bf16 bf16x16;
typedef __attribute__((ext_vector_type(8)))  float  f32x8;
typedef int vint4 __attribute__((vector_size(16)));
typedef __attribute__((address_space(1))) vint4 gas_vint4;
typedef __attribute__((address_space(3))) vint4 las_vint4;

#if defined(__has_builtin)
#  if __has_builtin(__builtin_amdgcn_global_load_async_to_lds_b128)
#    define USE_ASYNC_LDS 1
#  endif
#endif

union FragBF { bf16x16 v; unsigned int u[8]; uint4 q[2]; };

__device__ __forceinline__ unsigned short f2bf(float f) {
    union { __bf16 h; unsigned short u; } cv;
    cv.h = (__bf16)f;                       // RTNE via native hw convert
    return cv.u;
}
__device__ __forceinline__ unsigned int pack2bf(float a, float b) {
    union { __bf16 h[2]; unsigned int u; } cv;
    cv.h[0] = (__bf16)a;
    cv.h[1] = (__bf16)b;
    return cv.u;
}

// A fragment 16x32 bf16 from row-major LDS tile (stride LS).
// ISA: lane(0-15)=M, hf=lane>>4; v0..3 = K hf*8..hf*8+7 (contiguous), v4..7 = K 16+hf*8.. (contiguous)
__device__ __forceinline__ bf16x16 load_frag_a(const unsigned short* smem) {
    int lane = threadIdx.x & 31;
    int m = lane & 15, hf = lane >> 4;
    FragBF f;
    f.q[0] = *(const uint4*)(smem + m * LS + hf * 8);
    f.q[1] = *(const uint4*)(smem + m * LS + hf * 8 + 16);
    return f.v;
}
// B fragment 32x16 (KxN) from LDS stored transposed [n][k] (stride LS).
// ISA: N=lane&15, hf=lane>>4; v0..3 = K hf*8..hf*8+7, v4..7 = K 16+hf*8.. (same as A)
__device__ __forceinline__ bf16x16 load_frag_b(const unsigned short* smemT) {
    int lane = threadIdx.x & 31;
    int n = lane & 15, kb = (lane >> 4) * 8;
    FragBF f;
    f.q[0] = *(const uint4*)(smemT + n * LS + kb);
    f.q[1] = *(const uint4*)(smemT + n * LS + kb + 16);
    return f.v;
}
__device__ __forceinline__ f32x8 wmma_bf16(bf16x16 a, bf16x16 b, f32x8 c) {
    return __builtin_amdgcn_wmma_f32_16x16x32_bf16(false, a, false, b, (short)0, c, false, false);
}

// ---------------- zero output ----------------
__global__ __launch_bounds__(256) void zero_kernel(float* __restrict__ out) {
    size_t i = ((size_t)blockIdx.x * 256 + threadIdx.x) * 4;
    *(float4*)(out + i) = make_float4(0.f, 0.f, 0.f, 0.f);
}

// ---------------- router ----------------
__global__ __launch_bounds__(256) void router_kernel(const float* __restrict__ x,
                                                     const float* __restrict__ rw,
                                                     const float* __restrict__ rb,
                                                     float* __restrict__ comb) {
    int wid  = threadIdx.x >> 5;
    int lane = threadIdx.x & 31;
    int tok  = blockIdx.x * 8 + wid;
    const float* xr = x + (size_t)tok * HDIM;
    float acc[NEXP] = {};
    for (int i = lane; i < HDIM; i += 32) {
        float xv = xr[i];
#pragma unroll
        for (int e = 0; e < NEXP; ++e) acc[e] += xv * rw[e * HDIM + i];
    }
#pragma unroll
    for (int e = 0; e < NEXP; ++e)
#pragma unroll
        for (int off = 16; off > 0; off >>= 1)
            acc[e] += __shfl_xor(acc[e], off, 32);
    float score[NEXP], choice[NEXP];
#pragma unroll
    for (int e = 0; e < NEXP; ++e) {
        score[e]  = 1.f / (1.f + __expf(-acc[e]));
        choice[e] = score[e] + rb[e];
    }
    int i1 = 0;
#pragma unroll
    for (int e = 1; e < NEXP; ++e) if (choice[e] > choice[i1]) i1 = e;
    int i2 = (i1 == 0) ? 1 : 0;
#pragma unroll
    for (int e = 0; e < NEXP; ++e) if (e != i1 && choice[e] > choice[i2]) i2 = e;
    float denom = score[i1] + score[i2] + 1e-20f;
    if (lane < NEXP) {
        float w = (lane == i1) ? score[i1] : (lane == i2) ? score[i2] : 0.f;
        comb[(size_t)tok * NEXP + lane] = (w / denom) * 1.0f;   // ROUTED_SCALE = 1.0
    }
}

// ---------------- fused gate/up GEMM (64x128 tile, double-buffered) ----------------
__global__ __launch_bounds__(256)
void gateup_kernel(const float* __restrict__ x, const float* __restrict__ Wg,
                   const float* __restrict__ Wu, const float* __restrict__ comb,
                   int expert, unsigned short* __restrict__ hbuf) {
    __shared__ unsigned short Xs[2][64 * LS];
    __shared__ unsigned short Gs[2][128 * LS];   // transposed [n][k], k-pairs packed
    __shared__ unsigned short Us[2][128 * LS];
    const int tid = threadIdx.x;
    const int n0  = blockIdx.x * 128;
    const int t0  = blockIdx.y * 64;
    const int lane = tid & 31, wid = tid >> 5;
    const int wm = (wid & 1) * 32;     // 2 M-groups of 32
    const int wn = (wid >> 1) * 32;    // 4 N-groups of 32
    f32x8 aG[2][2] = {};
    f32x8 aU[2][2] = {};

    auto stage = [&](int b, int k0) {
        unsigned short* Xb = Xs[b];
        unsigned short* Gb = Gs[b];
        unsigned short* Ub = Us[b];
#pragma unroll
        for (int rep = 0; rep < 2; ++rep) {          // X tile 64x32
            int q = tid + rep * 256;
            int row = q >> 3, kc = (q & 7) * 4;
            float4 v = *(const float4*)(x + (size_t)(t0 + row) * HDIM + k0 + kc);
            *(uint2*)(Xb + row * LS + kc) = make_uint2(pack2bf(v.x, v.y), pack2bf(v.z, v.w));
        }
#pragma unroll
        for (int rep = 0; rep < 2; ++rep) {          // W tiles 32x128, k-pair packed transpose
            int t = tid + rep * 256;                 // 512 tasks
            int kp = t >> 5;                         // k-pair 0..15
            int nc = (t & 31) * 4;                   // coalesced along n
            const float* gp = Wg + (size_t)(k0 + 2 * kp) * IDIM + n0 + nc;
            const float* up = Wu + (size_t)(k0 + 2 * kp) * IDIM + n0 + nc;
            float4 g0 = *(const float4*)gp;
            float4 g1 = *(const float4*)(gp + IDIM);
            float4 u0 = *(const float4*)up;
            float4 u1 = *(const float4*)(up + IDIM);
            *(unsigned int*)(Gb + (nc + 0) * LS + 2 * kp) = pack2bf(g0.x, g1.x);
            *(unsigned int*)(Gb + (nc + 1) * LS + 2 * kp) = pack2bf(g0.y, g1.y);
            *(unsigned int*)(Gb + (nc + 2) * LS + 2 * kp) = pack2bf(g0.z, g1.z);
            *(unsigned int*)(Gb + (nc + 3) * LS + 2 * kp) = pack2bf(g0.w, g1.w);
            *(unsigned int*)(Ub + (nc + 0) * LS + 2 * kp) = pack2bf(u0.x, u1.x);
            *(unsigned int*)(Ub + (nc + 1) * LS + 2 * kp) = pack2bf(u0.y, u1.y);
            *(unsigned int*)(Ub + (nc + 2) * LS + 2 * kp) = pack2bf(u0.z, u1.z);
            *(unsigned int*)(Ub + (nc + 3) * LS + 2 * kp) = pack2bf(u0.w, u1.w);
        }
    };

    const int NIT = HDIM / 32;
    stage(0, 0);
    for (int i = 0; i < NIT; ++i) {
        __syncthreads();
        if (i + 1 < NIT) stage((i + 1) & 1, (i + 1) * 32);
        const unsigned short* Xb = Xs[i & 1];
        const unsigned short* Gb = Gs[i & 1];
        const unsigned short* Ub = Us[i & 1];
        bf16x16 a0  = load_frag_a(Xb + (wm + 0)  * LS);
        bf16x16 a1  = load_frag_a(Xb + (wm + 16) * LS);
        bf16x16 bg0 = load_frag_b(Gb + (wn + 0)  * LS);
        bf16x16 bg1 = load_frag_b(Gb + (wn + 16) * LS);
        bf16x16 bu0 = load_frag_b(Ub + (wn + 0)  * LS);
        bf16x16 bu1 = load_frag_b(Ub + (wn + 16) * LS);
        aG[0][0] = wmma_bf16(a0, bg0, aG[0][0]);
        aG[0][1] = wmma_bf16(a0, bg1, aG[0][1]);
        aG[1][0] = wmma_bf16(a1, bg0, aG[1][0]);
        aG[1][1] = wmma_bf16(a1, bg1, aG[1][1]);
        aU[0][0] = wmma_bf16(a0, bu0, aU[0][0]);
        aU[0][1] = wmma_bf16(a0, bu1, aU[0][1]);
        aU[1][0] = wmma_bf16(a1, bu0, aU[1][0]);
        aU[1][1] = wmma_bf16(a1, bu1, aU[1][1]);
    }
    int nn = lane & 15, hf = lane >> 4;
#pragma unroll
    for (int mt = 0; mt < 2; ++mt)
#pragma unroll
    for (int r = 0; r < 8; ++r) {
        int tok = t0 + wm + mt * 16 + hf * 8 + r;
        float w = comb ? comb[(size_t)tok * NEXP + expert] : 1.0f;
#pragma unroll
        for (int nt = 0; nt < 2; ++nt) {
            float g = aG[mt][nt][r], u = aU[mt][nt][r];
            float h = g * __builtin_amdgcn_rcpf(1.f + __expf(-g)) * u * w;
            hbuf[(size_t)tok * IDIM + n0 + wn + nt * 16 + nn] = f2bf(h);
        }
    }
}

// ---------------- down GEMM (128x64 tile, double-buffered, async h staging) ----------------
__global__ __launch_bounds__(256)
void down_kernel(const unsigned short* __restrict__ hbuf, const float* __restrict__ Wd,
                 float* __restrict__ out) {
    __shared__ unsigned short Hs[2][128 * LS];
    __shared__ unsigned short Ds[2][64 * LS];
    const int tid = threadIdx.x;
    const int n0  = blockIdx.x * 64;
    const int t0  = blockIdx.y * 128;
    const int lane = tid & 31, wid = tid >> 5;
    const int wm = (wid & 3) * 32;    // 4 M-groups of 32
    const int wn = (wid >> 2) * 32;   // 2 N-groups of 32
    f32x8 acc[2][2] = {};

    auto stage = [&](int b, int k0) {
        unsigned short* Hb = Hs[b];
        unsigned short* Db = Ds[b];
#pragma unroll
        for (int rep = 0; rep < 2; ++rep) {          // h tile 128x32 bf16: 512 x 16B chunks
            int q = tid + rep * 256;
            int row = q >> 2, kc = (q & 3) * 8;
            const unsigned short* src = hbuf + (size_t)(t0 + row) * IDIM + k0 + kc;
            unsigned short* dst = Hb + row * LS + kc;
#if USE_ASYNC_LDS
            __builtin_amdgcn_global_load_async_to_lds_b128(
                (gas_vint4*)const_cast<unsigned short*>(src),
                (las_vint4*)dst, 0, 0);
#else
            *(uint4*)dst = *(const uint4*)src;
#endif
        }
        {                                            // Wd tile 32x64, k-pair packed transpose
            int kp = tid >> 4;                       // 0..15
            int nc = (tid & 15) * 4;
            const float* dp = Wd + (size_t)(k0 + 2 * kp) * HDIM + n0 + nc;
            float4 d0 = *(const float4*)dp;
            float4 d1 = *(const float4*)(dp + HDIM);
            *(unsigned int*)(Db + (nc + 0) * LS + 2 * kp) = pack2bf(d0.x, d1.x);
            *(unsigned int*)(Db + (nc + 1) * LS + 2 * kp) = pack2bf(d0.y, d1.y);
            *(unsigned int*)(Db + (nc + 2) * LS + 2 * kp) = pack2bf(d0.z, d1.z);
            *(unsigned int*)(Db + (nc + 3) * LS + 2 * kp) = pack2bf(d0.w, d1.w);
        }
    };

    const int NIT = IDIM / 32;
    stage(0, 0);
    for (int i = 0; i < NIT; ++i) {
#if USE_ASYNC_LDS
#  if __has_builtin(__builtin_amdgcn_s_wait_asynccnt)
        __builtin_amdgcn_s_wait_asynccnt(0);
#  else
        asm volatile("s_wait_asynccnt 0x0" ::: "memory");
#  endif
#endif
        __syncthreads();
        if (i + 1 < NIT) stage((i + 1) & 1, (i + 1) * 32);
        const unsigned short* Hb = Hs[i & 1];
        const unsigned short* Db = Ds[i & 1];
        bf16x16 a0 = load_frag_a(Hb + (wm + 0)  * LS);
        bf16x16 a1 = load_frag_a(Hb + (wm + 16) * LS);
        bf16x16 b0 = load_frag_b(Db + (wn + 0)  * LS);
        bf16x16 b1 = load_frag_b(Db + (wn + 16) * LS);
        acc[0][0] = wmma_bf16(a0, b0, acc[0][0]);
        acc[0][1] = wmma_bf16(a0, b1, acc[0][1]);
        acc[1][0] = wmma_bf16(a1, b0, acc[1][0]);
        acc[1][1] = wmma_bf16(a1, b1, acc[1][1]);
    }
    int nn = lane & 15, hf = lane >> 4;
#pragma unroll
    for (int mt = 0; mt < 2; ++mt)
#pragma unroll
    for (int r = 0; r < 8; ++r) {
        int tok = t0 + wm + mt * 16 + hf * 8 + r;
        size_t o = (size_t)tok * HDIM + n0 + wn + nn;
        out[o]      += acc[mt][0][r];
        out[o + 16] += acc[mt][1][r];
    }
}

extern "C" void kernel_launch(void* const* d_in, const int* in_sizes, int n_in,
                              void* d_out, int out_size, void* d_ws, size_t ws_size,
                              hipStream_t stream) {
    (void)in_sizes; (void)n_in; (void)ws_size;
    const float* x   = (const float*)d_in[0];
    const float* rw  = (const float*)d_in[1];
    const float* rb  = (const float*)d_in[2];
    const float* Wg  = (const float*)d_in[3];
    const float* Wu  = (const float*)d_in[4];
    const float* Wd  = (const float*)d_in[5];
    const float* Wsg = (const float*)d_in[6];
    const float* Wsu = (const float*)d_in[7];
    const float* Wsd = (const float*)d_in[8];
    float* out = (float*)d_out;

    float*          comb = (float*)d_ws;                                         // T*E f32
    unsigned short* hbuf = (unsigned short*)((char*)d_ws + (size_t)TTOK * NEXP * 4); // T*I bf16

    zero_kernel<<<dim3(out_size / (256 * 4)), 256, 0, stream>>>(out);
    router_kernel<<<dim3(TTOK / 8), 256, 0, stream>>>(x, rw, rb, comb);

    dim3 gGU(IDIM / 128, TTOK / 64);   // 16 x 64
    dim3 gDN(HDIM / 64,  TTOK / 128);  // 16 x 32
    for (int e = 0; e < NEXP; ++e) {
        gateup_kernel<<<gGU, 256, 0, stream>>>(x, Wg + (size_t)e * HDIM * IDIM,
                                               Wu + (size_t)e * HDIM * IDIM, comb, e, hbuf);
        down_kernel<<<gDN, 256, 0, stream>>>(hbuf, Wd + (size_t)e * IDIM * HDIM, out);
    }
    gateup_kernel<<<gGU, 256, 0, stream>>>(x, Wsg, Wsu, nullptr, 0, hbuf);
    down_kernel<<<gDN, 256, 0, stream>>>(hbuf, Wsd, out);
}
